// S6Layer_21663815041548
// MI455X (gfx1250) — compile-verified, matrix operand order checked by reading, not run on hardware
//
#include <hip/hip_runtime.h>
#include <hip/hip_bf16.h>

typedef __attribute__((ext_vector_type(16))) _Float16 v16h;
typedef __attribute__((ext_vector_type(8)))  _Float16 v8h;
typedef __attribute__((ext_vector_type(4)))  _Float16 v4h;
typedef __attribute__((ext_vector_type(8)))  float    v8f;
typedef __attribute__((ext_vector_type(4)))  float    v4f;

#define B_SZ     2
#define L_SEQ    1024
#define D_MODEL  1024
#define D_STATE  16
#define D_MAMBA  2048
#define DT_RANK  64
#define XDBL_W   (DT_RANK + 2 * D_STATE)   // 96

#define TILE_M 128
#define TILE_N 128
#define TILE_K 32
#define LDS_STRIDE 48   // halves; 96B pitch keeps v8h (b128) loads 16B aligned

#define NCHUNK 16
#define LCHUNK (L_SEQ / NCHUNK)            // 64

// ---------------------------------------------------------------------------
// Staging helpers: global fp32 -> registers, registers -> f16 LDS
// ---------------------------------------------------------------------------
__device__ __forceinline__ void stage_load(
    const float* __restrict__ A, const float* __restrict__ B,
    int N, int lda, int ldb, int mBase, int nBase, int k0, int tid, bool fullN,
    v4f aReg[4], float bReg[16])
{
  #pragma unroll
  for (int i = 0; i < 4; ++i) {            // 128*32/4 = 1024 float4s
    int idx = tid + 256 * i;
    int row = idx >> 3;
    int c4  = idx & 7;
    aReg[i] = *(const v4f*)(A + (size_t)(mBase + row) * lda + k0 + c4 * 4);
  }
  if (fullN) {
    #pragma unroll
    for (int i = 0; i < 4; ++i) {          // 32*128/4 = 1024 float4s
      int idx  = tid + 256 * i;
      int krow = idx >> 5;
      int c4   = idx & 31;
      v4f bv = *(const v4f*)(B + (size_t)(k0 + krow) * ldb + nBase + c4 * 4);
      bReg[i * 4 + 0] = bv.x; bReg[i * 4 + 1] = bv.y;
      bReg[i * 4 + 2] = bv.z; bReg[i * 4 + 3] = bv.w;
    }
  } else {                                 // ragged N (N=96 GEMM): guarded scalars
    #pragma unroll
    for (int i = 0; i < 16; ++i) {         // 4096 elements
      int idx  = tid + 256 * i;
      int krow = idx >> 7;
      int n    = idx & 127;
      bReg[i] = (nBase + n < N) ? B[(size_t)(k0 + krow) * ldb + nBase + n] : 0.0f;
    }
  }
}

__device__ __forceinline__ void stage_store(
    _Float16* lA, _Float16* lBT, int tid, bool fullN,
    const v4f aReg[4], const float bReg[16])
{
  #pragma unroll
  for (int i = 0; i < 4; ++i) {
    int idx = tid + 256 * i;
    int row = idx >> 3;
    int c4  = idx & 7;
    v4h hv;
    hv.x = (_Float16)aReg[i].x; hv.y = (_Float16)aReg[i].y;
    hv.z = (_Float16)aReg[i].z; hv.w = (_Float16)aReg[i].w;
    *(v4h*)(lA + row * LDS_STRIDE + c4 * 4) = hv;
  }
  if (fullN) {
    #pragma unroll
    for (int i = 0; i < 4; ++i) {
      int idx  = tid + 256 * i;
      int krow = idx >> 5;
      int c4   = idx & 31;
      lBT[(c4 * 4 + 0) * LDS_STRIDE + krow] = (_Float16)bReg[i * 4 + 0];
      lBT[(c4 * 4 + 1) * LDS_STRIDE + krow] = (_Float16)bReg[i * 4 + 1];
      lBT[(c4 * 4 + 2) * LDS_STRIDE + krow] = (_Float16)bReg[i * 4 + 2];
      lBT[(c4 * 4 + 3) * LDS_STRIDE + krow] = (_Float16)bReg[i * 4 + 3];
    }
  } else {
    #pragma unroll
    for (int i = 0; i < 16; ++i) {
      int idx  = tid + 256 * i;
      int krow = idx >> 7;
      int n    = idx & 127;
      lBT[n * LDS_STRIDE + krow] = (_Float16)bReg[i];
    }
  }
}

// ---------------------------------------------------------------------------
// WMMA GEMM: C[M,N] = A[M,K] @ B[K,N]; fp32 in/out, f16 multiplicands, f32 acc.
// 128x128 block tile, double-buffered LDS, 8 waves each computing 32x64.
// All 6 fragments loaded into distinct registers first, then an 8-WMMA burst,
// so dscnt waits can be staggered instead of serialized.
// epilogue: 0 = none, 1 = softplus(x + bias[col])
// ---------------------------------------------------------------------------
__global__ __launch_bounds__(256) void wmma_gemm_f32(
    const float* __restrict__ A, const float* __restrict__ B, float* __restrict__ C,
    int M, int N, int K, int lda, int ldb, int ldc,
    const float* __restrict__ bias, int epilogue)
{
  __shared__ _Float16 ldsA [2][TILE_M * LDS_STRIDE];
  __shared__ _Float16 ldsBT[2][TILE_N * LDS_STRIDE];   // transposed: [n][k]

  const int tid  = threadIdx.x;
  const int wave = tid >> 5;
  const int lane = tid & 31;
  const int mBase = blockIdx.y * TILE_M;
  const int nBase = blockIdx.x * TILE_N;
  const int wm = (wave >> 1) * 32;           // wave row origin: 0/32/64/96
  const int wn = (wave & 1) * 64;            // wave col origin: 0/64
  const int frag_mn = lane & 15;
  const int kb      = (lane < 16) ? 0 : 8;   // K sub-block per ISA 16b layout
  const bool fullN  = (nBase + TILE_N <= N);

  v8f acc[2][4];
  #pragma unroll
  for (int im = 0; im < 2; ++im)
    #pragma unroll
    for (int nt = 0; nt < 4; ++nt) acc[im][nt] = (v8f)0.0f;

  v4f aReg[4];
  float bReg[16];

  const int kSteps = K / TILE_K;

  stage_load(A, B, N, lda, ldb, mBase, nBase, 0, tid, fullN, aReg, bReg);
  stage_store(ldsA[0], ldsBT[0], tid, fullN, aReg, bReg);
  __syncthreads();

  for (int ks = 0; ks < kSteps; ++ks) {
    const int cur = ks & 1;
    const int nxt = cur ^ 1;

    if (ks + 1 < kSteps)
      stage_load(A, B, N, lda, ldb, mBase, nBase, (ks + 1) * TILE_K, tid, fullN,
                 aReg, bReg);
    if (ks + 2 < kSteps) {
      __builtin_prefetch(A + (size_t)(mBase + (tid >> 1)) * lda + (ks + 2) * TILE_K, 0, 1);
      __builtin_prefetch(B + (size_t)((ks + 2) * TILE_K + (tid & 31)) * ldb + nBase, 0, 1);
    }

    // ---- load all fragments (distinct registers), then 8 WMMAs ----
    const _Float16* lA  = ldsA[cur];
    const _Float16* lBT = ldsBT[cur];

    v16h afrag[2], bfrag[4];
    #pragma unroll
    for (int im = 0; im < 2; ++im) {
      const _Float16* aRow = lA + (wm + im * 16 + frag_mn) * LDS_STRIDE;
      v8h lo = *(const v8h*)(aRow + kb);        // K = kb..kb+7
      v8h hi = *(const v8h*)(aRow + kb + 16);   // K = kb+16..kb+23
      #pragma unroll
      for (int e = 0; e < 8; ++e) { afrag[im][e] = lo[e]; afrag[im][e + 8] = hi[e]; }
    }
    #pragma unroll
    for (int nt = 0; nt < 4; ++nt) {
      const _Float16* bCol = lBT + (wn + nt * 16 + frag_mn) * LDS_STRIDE;
      v8h lo = *(const v8h*)(bCol + kb);
      v8h hi = *(const v8h*)(bCol + kb + 16);
      #pragma unroll
      for (int e = 0; e < 8; ++e) { bfrag[nt][e] = lo[e]; bfrag[nt][e + 8] = hi[e]; }
    }
    #pragma unroll
    for (int nt = 0; nt < 4; ++nt) {
      acc[0][nt] = __builtin_amdgcn_wmma_f32_16x16x32_f16(
          false, afrag[0], false, bfrag[nt], (short)0, acc[0][nt], false, false);
      acc[1][nt] = __builtin_amdgcn_wmma_f32_16x16x32_f16(
          false, afrag[1], false, bfrag[nt], (short)0, acc[1][nt], false, false);
    }

    if (ks + 1 < kSteps)
      stage_store(ldsA[nxt], ldsBT[nxt], tid, fullN, aReg, bReg);
    __syncthreads();
  }

  // ---- epilogue ----
  #pragma unroll
  for (int im = 0; im < 2; ++im) {
    const int rowTop = mBase + wm + im * 16 + ((lane < 16) ? 0 : 8);
    #pragma unroll
    for (int nt = 0; nt < 4; ++nt) {
      int col = nBase + wn + nt * 16 + (lane & 15);
      if (col < N) {
        float bia = (epilogue == 1) ? bias[col] : 0.0f;
        #pragma unroll
        for (int j = 0; j < 8; ++j) {
          float v = acc[im][nt][j];
          if (epilogue == 1) {
            v += bia;
            v = (v > 20.0f) ? v : __logf(1.0f + __expf(v));   // stable softplus
          }
          C[(size_t)(rowTop + j) * ldc + col] = v;
        }
      }
    }
  }
}

// ---------------------------------------------------------------------------
// Depthwise conv1d (k=4, pad left=1/right=2) + SiLU.  x_m = xz[:, 0:2048].
// ---------------------------------------------------------------------------
__global__ __launch_bounds__(256) void conv_silu_kernel(
    const float* __restrict__ xz, const float* __restrict__ conv_w,
    const float* __restrict__ conv_b, float* __restrict__ xc)
{
  int idx = blockIdx.x * 256 + threadIdx.x;
  if (idx >= B_SZ * L_SEQ * D_MAMBA) return;
  int d = idx % D_MAMBA;
  int l = (idx / D_MAMBA) % L_SEQ;
  int b = idx / (D_MAMBA * L_SEQ);
  float acc = conv_b[d];
  #pragma unroll
  for (int k = 0; k < 4; ++k) {
    int li = l - 1 + k;
    if (li >= 0 && li < L_SEQ)
      acc += conv_w[k * D_MAMBA + d] * xz[(size_t)(b * L_SEQ + li) * (2 * D_MAMBA) + d];
  }
  float sig = 1.0f / (1.0f + __expf(-acc));
  xc[(size_t)(b * L_SEQ + l) * D_MAMBA + d] = acc * sig;
}

// ---------------------------------------------------------------------------
// Chunked parallel linear-recurrence scan:  s_l = dA_l * s_{l-1} + dB_l*u_l
// Pass 1: per (b, chunk, d): P[s] = prod(dA), S[s] = local scan with s0 = 0.
// Summaries stored [b][chunk][s][d] so stores coalesce over d.
// ---------------------------------------------------------------------------
__global__ __launch_bounds__(128) void scan_pass1(
    const float* __restrict__ dt, const float* __restrict__ xc,
    const float* __restrict__ x_dbl, const float* __restrict__ A_log,
    float* __restrict__ chunkP, float* __restrict__ chunkS)
{
  const int d = blockIdx.x * 128 + threadIdx.x;
  const int c = blockIdx.y;
  const int b = blockIdx.z;

  float a_row[D_STATE], P[D_STATE], st[D_STATE];
  #pragma unroll
  for (int s = 0; s < D_STATE; ++s) {
    a_row[s] = -__expf(A_log[(size_t)d * D_STATE + s]);
    P[s] = 1.0f; st[s] = 0.0f;
  }
  const float* bcBase = x_dbl + (size_t)b * L_SEQ * XDBL_W + DT_RANK;

  for (int l = c * LCHUNK; l < (c + 1) * LCHUNK; ++l) {
    const size_t rl = (size_t)(b * L_SEQ + l);
    const v4f* bc4 = (const v4f*)(bcBase + (size_t)l * XDBL_W);
    v4f q[4];
    #pragma unroll
    for (int i = 0; i < 4; ++i) q[i] = bc4[i];        // Bc only
    const float* Bv = (const float*)q;

    const float dtv = dt[rl * D_MAMBA + d];
    const float u   = xc[rl * D_MAMBA + d];
    #pragma unroll
    for (int s = 0; s < D_STATE; ++s) {
      float dA = __expf(dtv * a_row[s]);
      P[s]  = P[s] * dA;
      st[s] = dA * st[s] + (dtv * Bv[s]) * u;
    }
  }
  const size_t base = ((size_t)(b * NCHUNK + c) * D_STATE) * D_MAMBA + d;
  #pragma unroll
  for (int s = 0; s < D_STATE; ++s) {
    chunkP[base + (size_t)s * D_MAMBA] = P[s];
    chunkS[base + (size_t)s * D_MAMBA] = st[s];
  }
}

// Pass 2: per (b,d,s), serial combine over the 16 chunks (tiny):
//   I_0 = 0;  I_{c+1} = P_c * I_c + S_c
__global__ __launch_bounds__(256) void scan_combine(
    const float* __restrict__ chunkP, const float* __restrict__ chunkS,
    float* __restrict__ chunkI)
{
  int idx = blockIdx.x * 256 + threadIdx.x;          // over B * D_STATE * D_MAMBA
  if (idx >= B_SZ * D_STATE * D_MAMBA) return;
  int d = idx % D_MAMBA;
  int s = (idx / D_MAMBA) % D_STATE;
  int b = idx / (D_MAMBA * D_STATE);
  float carry = 0.0f;
  for (int c = 0; c < NCHUNK; ++c) {
    size_t off = ((size_t)(b * NCHUNK + c) * D_STATE + s) * D_MAMBA + d;
    chunkI[off] = carry;
    carry = chunkP[off] * carry + chunkS[off];
  }
}

// Pass 3: re-run each chunk from its true initial state; emit gated outputs.
// Fuses  y += u*D  and the SiLU(z) gate.
__global__ __launch_bounds__(128) void scan_pass3(
    const float* __restrict__ dt,   const float* __restrict__ xc,
    const float* __restrict__ x_dbl, const float* __restrict__ xz,
    const float* __restrict__ A_log, const float* __restrict__ Dp,
    const float* __restrict__ chunkI, float* __restrict__ y)
{
  const int d = blockIdx.x * 128 + threadIdx.x;
  const int c = blockIdx.y;
  const int b = blockIdx.z;

  float a_row[D_STATE], st[D_STATE];
  const size_t ibase = ((size_t)(b * NCHUNK + c) * D_STATE) * D_MAMBA + d;
  #pragma unroll
  for (int s = 0; s < D_STATE; ++s) {
    a_row[s] = -__expf(A_log[(size_t)d * D_STATE + s]);
    st[s] = chunkI[ibase + (size_t)s * D_MAMBA];
  }
  const float Dd = Dp[d];
  const float* bcBase = x_dbl + (size_t)b * L_SEQ * XDBL_W + DT_RANK;

  for (int l = c * LCHUNK; l < (c + 1) * LCHUNK; ++l) {
    const size_t rl = (size_t)(b * L_SEQ + l);
    const v4f* bc4 = (const v4f*)(bcBase + (size_t)l * XDBL_W);
    v4f q[8];
    #pragma unroll
    for (int i = 0; i < 8; ++i) q[i] = bc4[i];        // Bc | Cc
    const float* bc = (const float*)q;

    const float dtv = dt[rl * D_MAMBA + d];
    const float u   = xc[rl * D_MAMBA + d];
    const float zv  = xz[rl * (2 * D_MAMBA) + D_MAMBA + d];

    float yv = 0.0f;
    #pragma unroll
    for (int s = 0; s < D_STATE; ++s) {
      float dA = __expf(dtv * a_row[s]);
      st[s] = dA * st[s] + (dtv * bc[s]) * u;
      yv += st[s] * bc[D_STATE + s];
    }
    yv += u * Dd;
    float sig = 1.0f / (1.0f + __expf(-zv));
    y[rl * D_MAMBA + d] = yv * (zv * sig);
  }
}

// ---------------------------------------------------------------------------
extern "C" void kernel_launch(void* const* d_in, const int* in_sizes, int n_in,
                              void* d_out, int out_size, void* d_ws, size_t ws_size,
                              hipStream_t stream) {
  const float* x      = (const float*)d_in[0];
  const float* W_in   = (const float*)d_in[1];
  const float* conv_w = (const float*)d_in[2];
  const float* conv_b = (const float*)d_in[3];
  const float* A_log  = (const float*)d_in[4];
  const float* Dp     = (const float*)d_in[5];
  const float* W_x    = (const float*)d_in[6];
  const float* W_dt   = (const float*)d_in[7];
  const float* b_dt   = (const float*)d_in[8];
  const float* W_out  = (const float*)d_in[9];
  float* out = (float*)d_out;

  const int M = B_SZ * L_SEQ;                       // 2048 rows everywhere
  const size_t SUMSZ = (size_t)B_SZ * NCHUNK * D_STATE * D_MAMBA;  // 1M floats

  float* ws     = (float*)d_ws;
  float* xz     = ws;                                // [M, 4096]
  float* xc     = xz     + (size_t)M * 2 * D_MAMBA;  // [M, 2048]
  float* x_dbl  = xc     + (size_t)M * D_MAMBA;      // [M, 96]
  float* dt     = x_dbl  + (size_t)M * XDBL_W;       // [M, 2048]
  float* ybuf   = dt     + (size_t)M * D_MAMBA;      // [M, 2048]
  float* chunkP = ybuf   + (size_t)M * D_MAMBA;      // [B,16,16,dm]
  float* chunkS = chunkP + SUMSZ;
  float* chunkI = chunkS + SUMSZ;

  // 1) xz = x @ W_in                      [2048,1024]x[1024,4096]
  wmma_gemm_f32<<<dim3((2 * D_MAMBA) / TILE_N, M / TILE_M), 256, 0, stream>>>(
      x, W_in, xz, M, 2 * D_MAMBA, D_MODEL, D_MODEL, 2 * D_MAMBA, 2 * D_MAMBA,
      nullptr, 0);

  // 2) depthwise conv + SiLU -> xc
  conv_silu_kernel<<<(B_SZ * L_SEQ * D_MAMBA) / 256, 256, 0, stream>>>(
      xz, conv_w, conv_b, xc);

  // 3) x_dbl = xc @ W_x                   [2048,2048]x[2048,96]  (ragged N)
  wmma_gemm_f32<<<dim3((XDBL_W + TILE_N - 1) / TILE_N, M / TILE_M), 256, 0, stream>>>(
      xc, W_x, x_dbl, M, XDBL_W, D_MAMBA, D_MAMBA, XDBL_W, XDBL_W,
      nullptr, 0);

  // 4) dt = softplus(x_dbl[:, :64] @ W_dt + b_dt)   [2048,64]x[64,2048]
  wmma_gemm_f32<<<dim3(D_MAMBA / TILE_N, M / TILE_M), 256, 0, stream>>>(
      x_dbl, W_dt, dt, M, D_MAMBA, DT_RANK, XDBL_W, D_MAMBA, D_MAMBA,
      b_dt, 1);

  // 5) chunked parallel selective scan -> ybuf
  scan_pass1<<<dim3(D_MAMBA / 128, NCHUNK, B_SZ), 128, 0, stream>>>(
      dt, xc, x_dbl, A_log, chunkP, chunkS);
  scan_combine<<<(B_SZ * D_STATE * D_MAMBA) / 256, 256, 0, stream>>>(
      chunkP, chunkS, chunkI);
  scan_pass3<<<dim3(D_MAMBA / 128, NCHUNK, B_SZ), 128, 0, stream>>>(
      dt, xc, x_dbl, xz, A_log, Dp, chunkI, ybuf);

  // 6) out = ybuf @ W_out                 [2048,2048]x[2048,1024]
  wmma_gemm_f32<<<dim3(D_MODEL / TILE_N, M / TILE_M), 256, 0, stream>>>(
      ybuf, W_out, out, M, D_MODEL, D_MAMBA, D_MAMBA, D_MODEL, D_MODEL,
      nullptr, 0);
}
